// GraphAttentionNetwork_352187318398
// MI455X (gfx1250) — matrix operand order repaired and checked by reading, not compiled, not run on hardware
//
#include <hip/hip_runtime.h>
#include <cmath>

typedef __attribute__((ext_vector_type(16))) __bf16        v16bf;
typedef __attribute__((ext_vector_type(8)))  float         v8f;
typedef __attribute__((ext_vector_type(4)))  unsigned int  u32x4;

union BPack { u32x4 u[2]; v16bf v; };

#define ALPHA 0.2f
#define BT    96      // B*T
#define NN    325
#define NPAD  336     // 21*16, padded neighbor count for packed bf16 buffers
#define FIN   64
#define FOUT  32
#define NH    4
#define HF    128     // NH*FOUT
#define ROW_TILES 21  // ceil(325/16)

static __device__ __forceinline__ v8f wmma_bf16(v16bf a, v16bf b, v8f c) {
  return __builtin_amdgcn_wmma_f32_16x16x32_bf16(
      false, a, false, b, (short)0, c, false, false);
}

// ---------------------------------------------------------------------------
// Repack W_heads [H][Fin][Fout] -> bf16 [H*Fout][Fin] (column-major for WMMA B)
// ---------------------------------------------------------------------------
__global__ void repack_Wheads(const float* __restrict__ W, __bf16* __restrict__ Bp) {
  int i = blockIdx.x * blockDim.x + threadIdx.x;   // i = c*FIN + k
  if (i < HF * FIN) {
    int c = i / FIN, k = i - c * FIN;
    int h = c / FOUT, o = c - h * FOUT;
    Bp[i] = (__bf16)W[((size_t)h * FIN + k) * FOUT + o];
  }
}

// W_out [HF][Fout] -> bf16 [Fout][HF]
__global__ void repack_Wout(const float* __restrict__ W, __bf16* __restrict__ Bp) {
  int i = blockIdx.x * blockDim.x + threadIdx.x;   // i = c*HF + k
  if (i < FOUT * HF) {
    int c = i / HF, k = i - c * HF;
    Bp[i] = (__bf16)W[(size_t)k * FOUT + c];
  }
}

// Zero the n = NN..NPAD-1 pad cells of a packed [BT][cols][NPAD] bf16 buffer.
__global__ void zero_pad(__bf16* __restrict__ p, int cols) {
  int i = blockIdx.x * blockDim.x + threadIdx.x;
  int total = BT * cols * (NPAD - NN);
  if (i < total) {
    int pad = i % (NPAD - NN);
    int rc  = i / (NPAD - NN);                     // bt*cols + col
    p[(size_t)rc * NPAD + NN + pad] = (__bf16)0.f;
  }
}

// ---------------------------------------------------------------------------
// WMMA GEMM: C[M x NCOLS] = A[M x K] @ B[K x NCOLS].
// A fp32 row-major; B prepacked bf16 column-major [col][K] (contiguous along K
// -> each B fragment is two aligned b128 loads). One wave per 16x(16*NTILES)
// C strip: NTILES accumulators share one A fragment (NTILES back-to-back
// v_wmma per k-step). Epilogue writes fp32 C and a transposed bf16 copy
// Cp[bt][col][n] (NPAD-padded) that feeds the attention kernel's B fragments.
// The (bt,n) split needs ONE division per lane: the 8 rows a lane covers are
// consecutive, so bt advances at most once (handled incrementally).
// ---------------------------------------------------------------------------
template <int K, int NCOLS, int NTILES>
__global__ void wmma_gemm(const float* __restrict__ A,
                          const __bf16* __restrict__ Bp,
                          float* __restrict__ C,
                          __bf16* __restrict__ Cp) {
  const int lane = threadIdx.x & 31;
  const int hl   = lane >> 4;
  const int lr   = lane & 15;
  const size_t row = (size_t)blockIdx.x * 16 + lr;
  const int    c0  = blockIdx.y * (16 * NTILES) + lr;

  const float*  arow = A + row * K;
  const __bf16* bptr[NTILES];
#pragma unroll
  for (int t = 0; t < NTILES; ++t)
    bptr[t] = Bp + (size_t)(c0 + 16 * t) * K + 8 * hl;

  v8f acc[NTILES];
#pragma unroll
  for (int t = 0; t < NTILES; ++t) acc[t] = (v8f){};

#pragma unroll
  for (int kb = 0; kb < K; kb += 32) {
    v16bf af;
#pragma unroll
    for (int e = 0; e < 8; ++e)                 // K = kb + 8*hl + e
      af[e] = (__bf16)arow[kb + 8 * hl + e];
#pragma unroll
    for (int e = 0; e < 8; ++e)                 // K = kb + 16 + 8*hl + e
      af[8 + e] = (__bf16)arow[kb + 16 + 8 * hl + e];
#pragma unroll
    for (int t = 0; t < NTILES; ++t) {
      BPack bp;
      bp.u[0] = *(const u32x4*)(bptr[t] + kb);
      bp.u[1] = *(const u32x4*)(bptr[t] + kb + 16);
      acc[t] = wmma_bf16(af, bp.v, acc[t]);
    }
  }

  // Epilogue: one idiv, incremental bt/n wrap.
  const int r2base = blockIdx.x * 16 + 8 * hl;  // rows r2base..r2base+7
  const int btb = r2base / NN;
  const int nb  = r2base - btb * NN;
#pragma unroll
  for (int i = 0; i < 8; ++i) {
    int bt = btb, n = nb + i;
    if (n >= NN) { bt += 1; n -= NN; }
    const size_t r2 = (size_t)r2base + i;
#pragma unroll
    for (int t = 0; t < NTILES; ++t) {
      const int c = c0 + 16 * t;
      float v = acc[t][i];
      C[r2 * NCOLS + c] = v;
      Cp[((size_t)bt * NCOLS + c) * NPAD + n] = (__bf16)v;
    }
  }
}

// ---------------------------------------------------------------------------
// Per-row attention scores: s1 = Wh_row . a[:Fout], s2 = Wh_row . a[Fout:]
// ---------------------------------------------------------------------------
__global__ void scores_kernel(const float* __restrict__ Wh, int ldw,
                              const float* __restrict__ a,
                              float* __restrict__ s1, float* __restrict__ s2,
                              int heads) {
  int idx = blockIdx.x * blockDim.x + threadIdx.x;
  int total = heads * BT * NN;
  if (idx >= total) return;
  int h = idx / (BT * NN);
  int r = idx - h * (BT * NN);
  const float* wr = Wh + (size_t)r * ldw + h * FOUT;
  const float* av = a + (size_t)h * 2 * FOUT;
  float v1 = 0.f, v2 = 0.f;
#pragma unroll
  for (int c = 0; c < FOUT; ++c) {
    v1 += wr[c] * av[c];
    v2 += wr[c] * av[FOUT + c];
  }
  s1[idx] = v1;
  s2[idx] = v2;
}

// ---------------------------------------------------------------------------
// Online softmax statistics per row over masked leakyrelu(s1[n]+s2[m]).
// One wave32 per (n, bt, head); shfl_xor tree combine. Stores max & 1/sum.
// ---------------------------------------------------------------------------
__global__ void stats_kernel(const float* __restrict__ s1,
                             const float* __restrict__ s2,
                             const int* __restrict__ adj,
                             float* __restrict__ mx, float* __restrict__ ism) {
  const int n = blockIdx.x, bt = blockIdx.y, h = blockIdx.z;
  const int lane = threadIdx.x;
  const size_t off = ((size_t)h * BT + bt) * NN;
  const float s1n = s1[off + n];
  float m = -3.0e38f, s = 0.f;
  for (int j = lane; j < NN; j += 32) {
    if (adj[(size_t)n * NN + j]) {
      float e = s1n + s2[off + j];
      e = e > 0.f ? e : ALPHA * e;
      if (e > m) { s = s * __expf(m - e) + 1.f; m = e; }
      else       { s += __expf(e - m); }
    }
  }
#pragma unroll
  for (int o = 16; o > 0; o >>= 1) {
    float om = __shfl_xor(m, o, 32);
    float os = __shfl_xor(s, o, 32);
    float nm = fmaxf(m, om);
    s = s * __expf(m - nm) + os * __expf(om - nm);
    m = nm;
  }
  if (lane == 0) {
    mx[off + n]  = m;
    ism[off + n] = s > 0.f ? 1.f / s : 0.f;
  }
}

// ---------------------------------------------------------------------------
// Fused attention aggregation + ELU: Out = elu(softmax(masked logits) @ Wh).
// One wave per (16-row tile, bt, head); the 16x32 probability A-fragment is
// built in registers once and fed to TWO wmma accumulators (32 output cols).
// B fragments come from the packed bf16 [bt][col][n] buffer: two b128 loads.
// 10 unconditional full chunks + 1 clamped tail chunk (NN=325).
// CPACK = packed col count = output row stride (128 stage 1, 32 stage 2).
// ---------------------------------------------------------------------------
template <int CPACK>
__global__ void attn_kernel(const __bf16* __restrict__ Whp,
                            const float* __restrict__ s1,
                            const float* __restrict__ s2,
                            const float* __restrict__ mx,
                            const float* __restrict__ ism,
                            const int* __restrict__ adj,
                            float* __restrict__ Out) {
  const int head = blockIdx.z;
  const int bt   = blockIdx.y;
  const int n0   = blockIdx.x * 16;
  const int lane = threadIdx.x & 31;
  const int hl   = lane >> 4;
  const int lr   = lane & 15;
  const int cb   = head * FOUT;

  const int n  = n0 + lr;
  const int nc = n < NN ? n : NN - 1;            // clamp ragged last row tile
  const size_t off = ((size_t)head * BT + bt) * NN;
  const float s1n  = s1[off + nc];
  const float mxn  = mx[off + nc];
  const float ismn = ism[off + nc];
  const int*   adjRow = adj + (size_t)nc * NN;
  const float* s2p    = s2 + off;

  const __bf16* b0 = Whp + ((size_t)bt * CPACK + cb + lr) * NPAD + 8 * hl;
  const __bf16* b1 = b0 + (size_t)16 * NPAD;

  v8f acc0 = {}, acc1 = {};
  int m0 = 0;
  for (; m0 + 32 <= NN; m0 += 32) {              // 10 full chunks, no clamping
    v16bf af;
#pragma unroll
    for (int g = 0; g < 2; ++g) {
      const int mb = m0 + (g ? 16 : 0) + 8 * hl;
#pragma unroll
      for (int e = 0; e < 8; ++e) {              // contiguous adj/s2 runs of 8
        int m = mb + e;
        float ev = s1n + s2p[m];
        ev = ev > 0.f ? ev : ALPHA * ev;
        float p = adjRow[m] ? __expf(ev - mxn) * ismn : 0.f;
        af[g * 8 + e] = (__bf16)p;
      }
    }
    BPack bp0, bp1;
    bp0.u[0] = *(const u32x4*)(b0 + m0);
    bp0.u[1] = *(const u32x4*)(b0 + m0 + 16);
    bp1.u[0] = *(const u32x4*)(b1 + m0);
    bp1.u[1] = *(const u32x4*)(b1 + m0 + 16);
    __builtin_prefetch(b0 + m0 + 32, 0, 1);      // global_prefetch_b8
    __builtin_prefetch(b1 + m0 + 32, 0, 1);
    acc0 = wmma_bf16(af, bp0.v, acc0);
    acc1 = wmma_bf16(af, bp1.v, acc1);
  }
  {                                              // tail chunk (m0 == 320)
    v16bf af;
#pragma unroll
    for (int g = 0; g < 2; ++g) {
      const int mb = m0 + (g ? 16 : 0) + 8 * hl;
#pragma unroll
      for (int e = 0; e < 8; ++e) {
        int m  = mb + e;
        int mc = m < NN ? m : NN - 1;            // stay in bounds
        float ev = s1n + s2p[mc];
        ev = ev > 0.f ? ev : ALPHA * ev;
        float p = (m < NN && adjRow[mc]) ? __expf(ev - mxn) * ismn : 0.f;
        af[g * 8 + e] = (__bf16)p;
      }
    }
    BPack bp0, bp1;                              // pad cells are zeroed
    bp0.u[0] = *(const u32x4*)(b0 + m0);
    bp0.u[1] = *(const u32x4*)(b0 + m0 + 16);
    bp1.u[0] = *(const u32x4*)(b1 + m0);
    bp1.u[1] = *(const u32x4*)(b1 + m0 + 16);
    acc0 = wmma_bf16(af, bp0.v, acc0);
    acc1 = wmma_bf16(af, bp1.v, acc1);
  }
#pragma unroll
  for (int i = 0; i < 8; ++i) {
    int row = n0 + i + 8 * hl;
    if (row < NN) {
      float v0 = acc0[i];
      float v1 = acc1[i];
      v0 = v0 > 0.f ? v0 : (__expf(v0) - 1.f);   // ELU epilogue
      v1 = v1 > 0.f ? v1 : (__expf(v1) - 1.f);
      float* orow = Out + ((size_t)bt * NN + row) * CPACK + cb + lr;
      orow[0]  = v0;
      orow[16] = v1;
    }
  }
}

// ---------------------------------------------------------------------------
extern "C" void kernel_launch(void* const* d_in, const int* in_sizes, int n_in,
                              void* d_out, int out_size, void* d_ws, size_t ws_size,
                              hipStream_t stream) {
  (void)in_sizes; (void)n_in; (void)out_size; (void)ws_size;
  const float* x       = (const float*)d_in[0];  // [96,325,64]
  const int*   adj     = (const int*)  d_in[1];  // [325,325]
  const float* W_heads = (const float*)d_in[2];  // [4,64,32]
  const float* a_heads = (const float*)d_in[3];  // [4,64,1]
  const float* W_out   = (const float*)d_in[4];  // [128,32]
  const float* a_out   = (const float*)d_in[5];  // [64,1]
  float* out = (float*)d_out;                    // [96,325,32]

  // 256-B aligned workspace carving (keeps all b128 accesses aligned).
  char* base = (char*)d_ws;
  size_t off = 0;
  auto carve = [&](size_t bytes) -> void* {
    void* p = base + off;
    off = (off + bytes + 255) & ~(size_t)255;
    return p;
  };
  float*  WhAll  = (float*) carve((size_t)BT * NN * HF * 4);    // 16 MB fp32
  float*  xcat   = (float*) carve((size_t)BT * NN * HF * 4);    // 16 MB fp32
  float*  Wh2    = (float*) carve((size_t)BT * NN * FOUT * 4);  //  4 MB fp32
  __bf16* WhbfP  = (__bf16*)carve((size_t)BT * HF * NPAD * 2);  // 8.3 MB bf16 packed
  __bf16* Wh2bfP = (__bf16*)carve((size_t)BT * FOUT * NPAD * 2);// 2.1 MB
  __bf16* WcatBf = (__bf16*)carve((size_t)HF * FIN * 2);
  __bf16* WoutBf = (__bf16*)carve((size_t)FOUT * HF * 2);
  float*  s1h  = (float*)carve((size_t)NH * BT * NN * 4);
  float*  s2h  = (float*)carve((size_t)NH * BT * NN * 4);
  float*  mxh  = (float*)carve((size_t)NH * BT * NN * 4);
  float*  ismh = (float*)carve((size_t)NH * BT * NN * 4);
  float*  s1o  = (float*)carve((size_t)BT * NN * 4);
  float*  s2o  = (float*)carve((size_t)BT * NN * 4);
  float*  mxo  = (float*)carve((size_t)BT * NN * 4);
  float*  ismo = (float*)carve((size_t)BT * NN * 4);

  // Weight repacks + pad zeroing (tiny).
  repack_Wheads<<<dim3((HF * FIN + 255) / 256), dim3(256), 0, stream>>>(W_heads, WcatBf);
  repack_Wout  <<<dim3((FOUT * HF + 255) / 256), dim3(256), 0, stream>>>(W_out, WoutBf);
  zero_pad<<<dim3((BT * HF   * (NPAD - NN) + 255) / 256), dim3(256), 0, stream>>>(WhbfP,  HF);
  zero_pad<<<dim3((BT * FOUT * (NPAD - NN) + 255) / 256), dim3(256), 0, stream>>>(Wh2bfP, FOUT);

  // Stage 1: all 4 heads. 4 col tiles per wave (64 cols), A fragment reused 4x.
  wmma_gemm<FIN, HF, 4><<<dim3(BT * NN / 16, HF / 64), dim3(32), 0, stream>>>(
      x, WcatBf, WhAll, WhbfP);
  {
    int total = NH * BT * NN;
    scores_kernel<<<dim3((total + 255) / 256), dim3(256), 0, stream>>>(
        WhAll, HF, a_heads, s1h, s2h, NH);
  }
  stats_kernel<<<dim3(NN, BT, NH), dim3(32), 0, stream>>>(s1h, s2h, adj, mxh, ismh);
  attn_kernel<HF><<<dim3(ROW_TILES, BT, NH), dim3(32), 0, stream>>>(
      WhbfP, s1h, s2h, mxh, ismh, adj, xcat);

  // Stage 2: output attention layer, final ELU fused.
  wmma_gemm<HF, FOUT, 2><<<dim3(BT * NN / 16, 1), dim3(32), 0, stream>>>(
      xcat, WoutBf, Wh2, Wh2bfP);
  {
    int total = BT * NN;
    scores_kernel<<<dim3((total + 255) / 256), dim3(256), 0, stream>>>(
        Wh2, FOUT, a_out, s1o, s2o, 1);
  }
  stats_kernel<<<dim3(NN, BT, 1), dim3(32), 0, stream>>>(s1o, s2o, adj, mxo, ismo);
  attn_kernel<FOUT><<<dim3(ROW_TILES, BT, 1), dim3(32), 0, stream>>>(
      Wh2bfP, s1o, s2o, mxo, ismo, adj, out);
}
